// TransformerBlock_35313221107766
// MI455X (gfx1250) — matrix operand order repaired and checked by reading, not compiled, not run on hardware
//
#include <hip/hip_runtime.h>
#include <hip/hip_bf16.h>

#define BB   4
#define TT   2048
#define DM   512
#define NH   8
#define DKK  64
#define DFFV 2048
#define BTV  (BB*TT)      // 8192
#define EPSV 1e-5f

typedef __attribute__((ext_vector_type(16))) __bf16        v16bf;
typedef __attribute__((ext_vector_type(8)))  float         v8f;
typedef __attribute__((ext_vector_type(8)))  unsigned short ushort8;

__device__ __forceinline__ unsigned short f2bf(float f) {
  unsigned int u = __float_as_uint(f);
  u += 0x7FFFu + ((u >> 16) & 1u);           // round-to-nearest-even
  return (unsigned short)(u >> 16);
}
__device__ __forceinline__ __bf16 bfbits(unsigned short u) {
  return __builtin_bit_cast(__bf16, u);
}
// two 16B loads -> one 16-element bf16 fragment
__device__ __forceinline__ v16bf load_frag(const unsigned short* p0, const unsigned short* p1) {
  ushort8 a = *(const ushort8*)p0;
  ushort8 b = *(const ushort8*)p1;
  v16bf r;
#pragma unroll
  for (int i = 0; i < 8; ++i) { r[i] = bfbits(a[i]); r[i + 8] = bfbits(b[i]); }
  return r;
}
__device__ __forceinline__ v8f wmma_bf16(v16bf a, v16bf b, v8f c) {
  return __builtin_amdgcn_wmma_f32_16x16x32_bf16(false, a, false, b, (short)0, c, false, false);
}

// ---------------- elementwise cast f32 -> bf16 ----------------
__global__ void k_cast(const float* __restrict__ src, unsigned short* __restrict__ dst, int n) {
  int i = blockIdx.x * 256 + threadIdx.x;
  if (i < n) dst[i] = f2bf(src[i]);
}

// ---------------- transpose-cast f32 [R,C] -> bf16 [C,R], grid.z batched ----------------
__global__ void k_tr(const float* __restrict__ src, unsigned short* __restrict__ dst, int R, int C) {
  long zo = (long)blockIdx.z * R * C;
  int i = blockIdx.x * 256 + threadIdx.x;
  if (i < R * C) {
    int r = i / C, c = i % C;
    dst[zo + (long)c * R + r] = f2bf(src[zo + i]);
  }
}

// ---------------- WMMA GEMM: C[M,N] = A(bf16 [M,K]) x Bt(bf16 [N,K])^T + bias ----------------
// grid.z batches weight blocks of size N*K (bias N, out M*N).
// Output modes are compile-time so the epilogue is branch-free straight-line code.
template <bool OUTF, bool OUTBF, bool OUTBFT>
__global__ void __launch_bounds__(256) k_gemm(
    const unsigned short* __restrict__ A, const unsigned short* __restrict__ Bt,
    const float* __restrict__ bias,
    float* __restrict__ outF, unsigned short* __restrict__ outBf,
    unsigned short* __restrict__ outBfT,
    int M, int N, int K) {
  const int lane = threadIdx.x & 31;
  const int wave = threadIdx.x >> 5;
  const int half = lane >> 4;
  const int ln   = lane & 15;
  const int z    = blockIdx.z;
  const long zoffO = (long)z * M * N;
  const float* biasZ = bias + (long)z * N;
  const unsigned short* Bz = Bt + (long)z * N * K;

  const int m0 = blockIdx.x * 128 + (wave >> 1) * 32;
  const int n0 = blockIdx.y * 128 + (wave & 1) * 64;
  if (n0 >= N) return;

  v8f acc[2][4];
#pragma unroll
  for (int i = 0; i < 2; ++i)
#pragma unroll
    for (int j = 0; j < 4; ++j)
#pragma unroll
      for (int r = 0; r < 8; ++r) acc[i][j][r] = 0.f;

  const int kbA = half * 16;   // A-fragment K base within 32-chunk
  const int kbB = half * 16;   // B-fragment K base within 32-chunk

  for (int kk = 0; kk < K; kk += 32) {
    v16bf af[2], bfv[4];
#pragma unroll
    for (int ms = 0; ms < 2; ++ms) {
      const unsigned short* ap = A + (long)(m0 + ms * 16 + ln) * K + kk;
      __builtin_prefetch(ap + 32, 0, 3);   // one K-step (64B) ahead, near-cache
      af[ms] = load_frag(ap + kbA, ap + 8 + kbA);
    }
#pragma unroll
    for (int ns = 0; ns < 4; ++ns) {
      const unsigned short* bp = Bz + (long)(n0 + ns * 16 + ln) * K + kk + kbB;
      bfv[ns] = load_frag(bp, bp + 8);
    }
#pragma unroll
    for (int ms = 0; ms < 2; ++ms)
#pragma unroll
      for (int ns = 0; ns < 4; ++ns)
        acc[ms][ns] = wmma_bf16(af[ms], bfv[ns], acc[ms][ns]);
  }

#pragma unroll
  for (int ms = 0; ms < 2; ++ms)
#pragma unroll
    for (int ns = 0; ns < 4; ++ns) {
      int c = n0 + ns * 16 + ln;
      float bb = biasZ[c];
#pragma unroll
      for (int r = 0; r < 8; ++r) {
        int m = m0 + ms * 16 + r + 8 * half;
        float val = acc[ms][ns][r] + bb;
        long idx = zoffO + (long)m * N + c;
        if (OUTF)   outF[idx]  = val;
        if (OUTBF)  outBf[idx] = f2bf(val);
        if (OUTBFT) outBfT[zoffO + (long)c * M + m] = f2bf(val);
      }
    }
}

// ---------------- flash attention: q,k bf16 [h][b*T+t][64]; vt bf16 [h][64][b*T+t] ----------------
// writes concatenated heads bf16 [b*T+t][512]
__global__ void __launch_bounds__(256) k_attn(
    const unsigned short* __restrict__ q, const unsigned short* __restrict__ kmat,
    const unsigned short* __restrict__ vt, unsigned short* __restrict__ catbf) {
  __shared__ __align__(16) unsigned short pbuf[8][16][32];
  const int lane = threadIdx.x & 31;
  const int wave = threadIdx.x >> 5;
  const int half = lane >> 4;
  const int ln   = lane & 15;
  const int h = blockIdx.y / BB;
  const int b = blockIdx.y % BB;
  const long qoff  = ((long)h * BTV + (long)b * TT) * DKK;
  const long vtoff = (long)h * DKK * BTV + (long)b * TT;
  const int qt0 = blockIdx.x * 128 + wave * 16;
  const int kbA = half * 16;

  v16bf aq[2];
#pragma unroll
  for (int kc = 0; kc < 2; ++kc) {
    const unsigned short* qp = q + qoff + (long)(qt0 + ln) * DKK + kc * 32;
    aq[kc] = load_frag(qp + kbA, qp + 8 + kbA);
  }

  v8f o[4];
  float mi[8], li[8];
#pragma unroll
  for (int nc = 0; nc < 4; ++nc)
#pragma unroll
    for (int r = 0; r < 8; ++r) o[nc][r] = 0.f;
#pragma unroll
  for (int r = 0; r < 8; ++r) { mi[r] = -1e30f; li[r] = 0.f; }

  const float scl = 0.125f;  // 1/sqrt(64)

  for (int kb = 0; kb < TT; kb += 32) {
    v8f s0, s1;
#pragma unroll
    for (int r = 0; r < 8; ++r) { s0[r] = 0.f; s1[r] = 0.f; }
#pragma unroll
    for (int kc = 0; kc < 2; ++kc) {
      const unsigned short* kp0 = kmat + qoff + (long)(kb + ln) * DKK + kc * 32 + half * 16;
      v16bf bk0 = load_frag(kp0, kp0 + 8);
      s0 = wmma_bf16(aq[kc], bk0, s0);
      const unsigned short* kp1 = kmat + qoff + (long)(kb + 16 + ln) * DKK + kc * 32 + half * 16;
      v16bf bk1 = load_frag(kp1, kp1 + 8);
      s1 = wmma_bf16(aq[kc], bk1, s1);
    }
#pragma unroll
    for (int r = 0; r < 8; ++r) {
      float v0 = s0[r] * scl, v1 = s1[r] * scl;
      float mx = fmaxf(v0, v1);
#pragma unroll
      for (int off = 1; off < 16; off <<= 1) mx = fmaxf(mx, __shfl_xor(mx, off, 32));
      float mnew = fmaxf(mi[r], mx);
      float corr = expf(mi[r] - mnew);
      float p0 = expf(v0 - mnew), p1 = expf(v1 - mnew);
      float rs = p0 + p1;
#pragma unroll
      for (int off = 1; off < 16; off <<= 1) rs += __shfl_xor(rs, off, 32);
      li[r] = li[r] * corr + rs;
      mi[r] = mnew;
#pragma unroll
      for (int nc = 0; nc < 4; ++nc) o[nc][r] *= corr;
      pbuf[wave][r + 8 * half][ln]      = f2bf(p0);
      pbuf[wave][r + 8 * half][16 + ln] = f2bf(p1);
    }
    __syncthreads();
    const unsigned short* pp = &pbuf[wave][ln][0];
    v16bf ap = load_frag(pp + kbA, pp + 8 + kbA);
#pragma unroll
    for (int nc = 0; nc < 4; ++nc) {
      const unsigned short* vp = vt + vtoff + (long)(nc * 16 + ln) * BTV + kb + half * 16;
      v16bf bv = load_frag(vp, vp + 8);
      o[nc] = wmma_bf16(ap, bv, o[nc]);
    }
    __syncthreads();
  }

#pragma unroll
  for (int nc = 0; nc < 4; ++nc)
#pragma unroll
    for (int r = 0; r < 8; ++r) {
      int t = qt0 + r + 8 * half;
      long row = (long)b * TT + t;
      int col = h * DKK + nc * 16 + ln;
      catbf[row * DM + col] = f2bf(o[nc][r] / li[r]);
    }
}

// ---------------- fused (residual-add -> LN -> [gelu] -> base + s*t) ----------------
// All optional paths are template params -> branch-free element loop.
template <bool RESID, bool GELU, bool BASE, bool OUTF, bool OUTBF>
__global__ void __launch_bounds__(256) k_ln(
    const float* __restrict__ a, const float* __restrict__ resid,
    const float* __restrict__ g, const float* __restrict__ bb,
    const float* __restrict__ base, const float* __restrict__ sptr,
    int N, float* __restrict__ outF, unsigned short* __restrict__ outBf) {
  const int row = blockIdx.x;
  const long roff = (long)row * N;
  const int tid = threadIdx.x;
  float sum = 0.f, sq = 0.f;
  for (int i = tid; i < N; i += 256) {
    float v = a[roff + i];
    if (RESID) v += resid[roff + i];
    sum += v; sq += v * v;
  }
  __shared__ float s_sum[8], s_sq[8];
  __shared__ float s_mu, s_rstd;
#pragma unroll
  for (int off = 16; off >= 1; off >>= 1) {
    sum += __shfl_xor(sum, off, 32);
    sq  += __shfl_xor(sq,  off, 32);
  }
  if ((tid & 31) == 0) { s_sum[tid >> 5] = sum; s_sq[tid >> 5] = sq; }
  __syncthreads();
  if (tid == 0) {
    float ts = 0.f, tq = 0.f;
    for (int w = 0; w < 8; ++w) { ts += s_sum[w]; tq += s_sq[w]; }
    float mu = ts / N;
    s_mu = mu;
    s_rstd = rsqrtf(tq / N - mu * mu + EPSV);
  }
  __syncthreads();
  float mu = s_mu, rstd = s_rstd;
  float s = sptr ? sptr[0] : 1.f;   // single uniform scalar load
  for (int i = tid; i < N; i += 256) {
    float v = a[roff + i];
    if (RESID) v += resid[roff + i];
    float t = (v - mu) * rstd * g[i] + bb[i];
    if (GELU) t = 0.5f * t * (1.f + erff(t * 0.70710678118654752f));
    float y = s * t;
    if (BASE) y += base[roff + i];
    if (OUTF)  outF[roff + i]  = y;
    if (OUTBF) outBf[roff + i] = f2bf(y);
  }
}

extern "C" void kernel_launch(void* const* d_in, const int* in_sizes, int n_in,
                              void* d_out, int out_size, void* d_ws, size_t ws_size,
                              hipStream_t stream) {
  (void)in_sizes; (void)n_in; (void)out_size; (void)ws_size;
  const float* x      = (const float*)d_in[0];
  const float* Wq     = (const float*)d_in[1];
  const float* bq     = (const float*)d_in[2];
  const float* Wk     = (const float*)d_in[3];
  const float* bk     = (const float*)d_in[4];
  const float* Wv     = (const float*)d_in[5];
  const float* bv     = (const float*)d_in[6];
  const float* Wo     = (const float*)d_in[7];
  const float* bo     = (const float*)d_in[8];
  const float* ln1_g  = (const float*)d_in[9];
  const float* ln1_b  = (const float*)d_in[10];
  const float* ff_W1  = (const float*)d_in[11];
  const float* ff_b1  = (const float*)d_in[12];
  const float* ff_lng = (const float*)d_in[13];
  const float* ff_lnb = (const float*)d_in[14];
  const float* ff_W2  = (const float*)d_in[15];
  const float* ff_b2  = (const float*)d_in[16];
  const float* ln2_g  = (const float*)d_in[17];
  const float* ln2_b  = (const float*)d_in[18];
  const float* alphaA = (const float*)d_in[19];
  const float* alphaB = (const float*)d_in[20];
  const float* p1     = (const float*)d_in[21];
  const float* s1_W1  = (const float*)d_in[22];
  const float* s1_b1  = (const float*)d_in[23];
  const float* s1_lng = (const float*)d_in[24];
  const float* s1_lnb = (const float*)d_in[25];
  const float* s1_W2  = (const float*)d_in[26];
  const float* s1_b2  = (const float*)d_in[27];
  const float* lns1_g = (const float*)d_in[28];
  const float* lns1_b = (const float*)d_in[29];
  const float* p2     = (const float*)d_in[30];
  const float* sub2_W = (const float*)d_in[31];
  const float* sub2_b = (const float*)d_in[32];
  const float* lns2_g = (const float*)d_in[33];
  const float* lns2_b = (const float*)d_in[34];
  float* outp = (float*)d_out;

  char* ws = (char*)d_ws;
  size_t off = 0;
  auto alloc = [&](size_t bytes) -> void* {
    void* p = ws + off;
    off += (bytes + 255) & ~(size_t)255;
    return p;
  };
  unsigned short* x_bf   = (unsigned short*)alloc((size_t)BTV * DM * 2);
  unsigned short* wq_t   = (unsigned short*)alloc((size_t)NH * DKK * DM * 2);
  unsigned short* wk_t   = (unsigned short*)alloc((size_t)NH * DKK * DM * 2);
  unsigned short* wv_t   = (unsigned short*)alloc((size_t)NH * DKK * DM * 2);
  unsigned short* wo_t   = (unsigned short*)alloc((size_t)DM * DM * 2);
  unsigned short* w1_t   = (unsigned short*)alloc((size_t)DFFV * DM * 2);
  unsigned short* w2_t   = (unsigned short*)alloc((size_t)DM * DFFV * 2);
  unsigned short* s1w1_t = (unsigned short*)alloc((size_t)DFFV * DM * 2);
  unsigned short* s1w2_t = (unsigned short*)alloc((size_t)DM * DFFV * 2);
  unsigned short* s2w_t  = (unsigned short*)alloc((size_t)DM * DM * 2);
  unsigned short* qb     = (unsigned short*)alloc((size_t)NH * BTV * DKK * 2);
  unsigned short* kb2    = (unsigned short*)alloc((size_t)NH * BTV * DKK * 2);
  unsigned short* vtb    = (unsigned short*)alloc((size_t)NH * BTV * DKK * 2);
  unsigned short* catb   = (unsigned short*)alloc((size_t)BTV * DM * 2);
  float*          f1     = (float*)alloc((size_t)BTV * DM * 4);    // attn / fft / sub outs (reused)
  float*          out1f  = (float*)alloc((size_t)BTV * DM * 4);
  unsigned short* out1bf = (unsigned short*)alloc((size_t)BTV * DM * 2);
  float*          x1f    = (float*)alloc((size_t)BTV * DFFV * 4);  // reused by sub1
  unsigned short* hbf    = (unsigned short*)alloc((size_t)BTV * DFFV * 2);

  dim3 blk(256);
  // --- precision casts / weight transposes ---
  k_cast<<<(BTV * DM) / 256, blk, 0, stream>>>(x, x_bf, BTV * DM);
  k_tr<<<dim3((DM * DKK + 255) / 256, 1, NH), blk, 0, stream>>>(Wq, wq_t, DM, DKK);
  k_tr<<<dim3((DM * DKK + 255) / 256, 1, NH), blk, 0, stream>>>(Wk, wk_t, DM, DKK);
  k_tr<<<dim3((DM * DKK + 255) / 256, 1, NH), blk, 0, stream>>>(Wv, wv_t, DM, DKK);
  k_tr<<<dim3((DM * DM + 255) / 256, 1, 1), blk, 0, stream>>>(Wo, wo_t, DM, DM);
  k_tr<<<dim3((DM * DFFV + 255) / 256, 1, 1), blk, 0, stream>>>(ff_W1, w1_t, DM, DFFV);
  k_tr<<<dim3((DFFV * DM + 255) / 256, 1, 1), blk, 0, stream>>>(ff_W2, w2_t, DFFV, DM);
  k_tr<<<dim3((DM * DFFV + 255) / 256, 1, 1), blk, 0, stream>>>(s1_W1, s1w1_t, DM, DFFV);
  k_tr<<<dim3((DFFV * DM + 255) / 256, 1, 1), blk, 0, stream>>>(s1_W2, s1w2_t, DFFV, DM);
  k_tr<<<dim3((DM * DM + 255) / 256, 1, 1), blk, 0, stream>>>(sub2_W, s2w_t, DM, DM);

  // --- QKV projections (per-head, grid.z = H); V stored transposed for flash PV ---
  dim3 gqkv(BTV / 128, 1, NH);
  k_gemm<false, true, false><<<gqkv, blk, 0, stream>>>(x_bf, wq_t, bq, nullptr, qb, nullptr, BTV, DKK, DM);
  k_gemm<false, true, false><<<gqkv, blk, 0, stream>>>(x_bf, wk_t, bk, nullptr, kb2, nullptr, BTV, DKK, DM);
  k_gemm<false, false, true><<<gqkv, blk, 0, stream>>>(x_bf, wv_t, bv, nullptr, nullptr, vtb, BTV, DKK, DM);

  // --- flash attention -> concatenated heads (bf16) ---
  k_attn<<<dim3(TT / 128, NH * BB), blk, 0, stream>>>(qb, kb2, vtb, catb);

  // --- output projection, ln1, out1 ---
  k_gemm<true, false, false><<<dim3(BTV / 128, DM / 128, 1), blk, 0, stream>>>(
      catb, wo_t, bo, f1, nullptr, nullptr, BTV, DM, DM);
  k_ln<true, false, true, true, true><<<BTV, blk, 0, stream>>>(
      f1, x, ln1_g, ln1_b, x, alphaA, DM, out1f, out1bf);

  // --- main FFN ---
  k_gemm<true, false, false><<<dim3(BTV / 128, DFFV / 128, 1), blk, 0, stream>>>(
      out1bf, w1_t, ff_b1, x1f, nullptr, nullptr, BTV, DFFV, DM);
  k_ln<false, true, false, false, true><<<BTV, blk, 0, stream>>>(
      x1f, nullptr, ff_lng, ff_lnb, nullptr, nullptr, DFFV, nullptr, hbf);
  k_gemm<true, false, false><<<dim3(BTV / 128, DM / 128, 1), blk, 0, stream>>>(
      hbf, w2_t, ff_b2, f1, nullptr, nullptr, BTV, DM, DFFV);
  // ln2 -> out2 written straight to d_out
  k_ln<true, false, true, true, false><<<BTV, blk, 0, stream>>>(
      f1, out1f, ln2_g, ln2_b, out1f, alphaB, DM, outp, nullptr);

  // --- sub-branch 1: FFN on x, then p1*gelu(LN(.)) accumulated into d_out ---
  k_gemm<true, false, false><<<dim3(BTV / 128, DFFV / 128, 1), blk, 0, stream>>>(
      x_bf, s1w1_t, s1_b1, x1f, nullptr, nullptr, BTV, DFFV, DM);
  k_ln<false, true, false, false, true><<<BTV, blk, 0, stream>>>(
      x1f, nullptr, s1_lng, s1_lnb, nullptr, nullptr, DFFV, nullptr, hbf);
  k_gemm<true, false, false><<<dim3(BTV / 128, DM / 128, 1), blk, 0, stream>>>(
      hbf, s1w2_t, s1_b2, f1, nullptr, nullptr, BTV, DM, DFFV);
  k_ln<false, true, true, true, false><<<BTV, blk, 0, stream>>>(
      f1, nullptr, lns1_g, lns1_b, outp, p1, DM, outp, nullptr);

  // --- sub-branch 2: x@W + b, p2*gelu(LN(.)) accumulated into d_out ---
  k_gemm<true, false, false><<<dim3(BTV / 128, DM / 128, 1), blk, 0, stream>>>(
      x_bf, s2w_t, sub2_b, f1, nullptr, nullptr, BTV, DM, DM);
  k_ln<false, true, true, true, false><<<BTV, blk, 0, stream>>>(
      f1, nullptr, lns2_g, lns2_b, outp, p2, DM, outp, nullptr);
}
